// Head_84576495993092
// MI455X (gfx1250) — matrix operand order repaired and checked by reading, not compiled, not run on hardware
//
#include <hip/hip_runtime.h>
#include <hip/hip_bf16.h>

typedef __attribute__((ext_vector_type(16))) _Float16 v16h;
typedef __attribute__((ext_vector_type(8)))  float    v8f;

#define T_DIM 64
#define C_DIM 192
#define H_DIM 32
#define XSTR  200   // padded LDS stride for x (f16 elements)

// ---------------- gravity term: grav[t] = -G / sum_n((t - tgt_n)^2 + 1e-6) ----
__global__ void grav_kernel(const float* __restrict__ tgt, float* __restrict__ grav, int n) {
    __shared__ float red[256];
    const float tf = (float)blockIdx.x;   // t = 0..63
    float acc = 0.0f;
    for (int i = threadIdx.x; i < n; i += 256) {
        float d = tf - tgt[i];
        acc += d * d + 1e-6f;
    }
    red[threadIdx.x] = acc;
    __syncthreads();
    for (int s = 128; s > 0; s >>= 1) {
        if ((int)threadIdx.x < s) red[threadIdx.x] += red[threadIdx.x + s];
        __syncthreads();
    }
    if (threadIdx.x == 0) grav[blockIdx.x] = -5.25f / red[0];
}

// ---------------- fragment loaders (contiguous k = 16*half + s in both A and B) --------
// A (16x32 f16): lane covers row; VGPR v: K = 16*half+2v (matches B)
__device__ __forceinline__ int a_koff(int v, int half) {
    return 16 * half + 2 * v;
}

__device__ __forceinline__ v16h load_a_h16(const _Float16* S, int ld, int row, int k0, int half) {
    const _Float16* p = S + row * ld + k0;
    v16h a;
#pragma unroll
    for (int v = 0; v < 8; ++v) {
        int kk = a_koff(v, half);
        a[2 * v]     = p[kk];
        a[2 * v + 1] = p[kk + 1];
    }
    return a;
}

__device__ __forceinline__ v16h load_a_f32(const float* S, int ld, int row, int k0, int half) {
    const float* p = S + row * ld + k0;
    v16h a;
#pragma unroll
    for (int v = 0; v < 8; ++v) {
        int kk = a_koff(v, half);
        a[2 * v]     = (_Float16)p[kk];
        a[2 * v + 1] = (_Float16)p[kk + 1];
    }
    return a;
}

// B (32x16 f16): N = n0+lw; VGPR v holds K = k0+16*half+2v (lo) / +1 (hi).
// Accessor is row-major in K: B(k,n) = S[(n0+lw)*ld + k]
__device__ __forceinline__ v16h load_b_h16(const _Float16* S, int ld, int n0, int k0, int lw, int half) {
    const _Float16* p = S + (n0 + lw) * ld + k0 + 16 * half;
    v16h b;
#pragma unroll
    for (int v = 0; v < 8; ++v) {
        b[2 * v]     = p[2 * v];
        b[2 * v + 1] = p[2 * v + 1];
    }
    return b;
}

// B fragment straight from global f32 weight matrix W (H x C, row-major):
// B(k,n) = W[n*C + k]
__device__ __forceinline__ v16h load_b_w(const float* W, int n0, int k0, int lw, int half) {
    const float* p = W + (n0 + lw) * C_DIM + k0 + 16 * half;
    v16h b;
#pragma unroll
    for (int v = 0; v < 8; ++v) {
        b[2 * v]     = (_Float16)p[2 * v];
        b[2 * v + 1] = (_Float16)p[2 * v + 1];
    }
    return b;
}

__device__ __forceinline__ v8f wmma16(v16h a, v16h b, v8f c) {
    return __builtin_amdgcn_wmma_f32_16x16x32_f16(false, a, false, b, (short)0, c, false, false);
}

// ---------------- fused attention head: one workgroup per batch element ------
__global__ void __launch_bounds__(256)
attn_kernel(const float* __restrict__ x,
            const float* __restrict__ Wq,
            const float* __restrict__ Wk,
            const float* __restrict__ Wv,
            const float* __restrict__ grav,
            float* __restrict__ out) {
    __shared__ _Float16 xs[T_DIM * XSTR];        // x tile, f16, padded stride
    __shared__ _Float16 qs[T_DIM * H_DIM];       // q, row-major
    __shared__ _Float16 ks[T_DIM * H_DIM];       // k, row-major
    __shared__ _Float16 vts[H_DIM * T_DIM];      // v transposed: vts[d*64 + s]
    __shared__ float    wsm[T_DIM * T_DIM];      // attention scores (f32)
    __shared__ float    gsh[T_DIM];              // gravity bias per row
    __shared__ float    red2[256];               // softmax reduction scratch

    const int tid  = threadIdx.x;
    const int wv   = tid >> 5;       // wave id 0..7 (wave32)
    const int lane = tid & 31;
    const int half = lane >> 4;
    const int lw   = lane & 15;
    const int b    = blockIdx.x;

    const float* xb = x + (size_t)b * (T_DIM * C_DIM);

    // ---- phase 0: stage x (f32 -> f16) into LDS; preload gravity bias ----
#pragma unroll
    for (int j = 0; j < (T_DIM * C_DIM) / 256; ++j) {
        int idx = j * 256 + tid;
        int row = idx / C_DIM;
        int col = idx - row * C_DIM;
        xs[row * XSTR + col] = (_Float16)xb[idx];
    }
    if (tid < T_DIM) gsh[tid] = grav[tid];
    __syncthreads();

    // ---- phase 1: projections q,k,v = x @ W^T  (24 tiles, 3 per wave) ----
    // (wv + 8p) & 7 == wv, so a wave's (tm,tn) is the same for all three
    // projections: load the 6 x A-fragments ONCE and reuse across q/k/v.
    {
        const int tm = wv >> 1;      // 0..3 (M tile)
        const int tn = wv & 1;       // 0..1 (N tile)
        v16h afr[6];
#pragma unroll
        for (int kk = 0; kk < 6; ++kk)
            afr[kk] = load_a_h16(xs, XSTR, tm * 16 + lw, kk * 32, half);

#pragma unroll
        for (int p = 0; p < 3; ++p) {
            const float* W = (p == 0) ? Wq : ((p == 1) ? Wk : Wv);
            v8f c = {};
#pragma unroll
            for (int kk = 0; kk < 6; ++kk)
                c = wmma16(afr[kk], load_b_w(W, tn * 16, kk * 32, lw, half), c);
            // scatter D: element (m = tm*16 + 8*half + i, n = tn*16 + lw)
            if (p == 2) {
#pragma unroll
                for (int i = 0; i < 8; ++i)
                    vts[(tn * 16 + lw) * T_DIM + (tm * 16 + 8 * half + i)] = (_Float16)c[i];
            } else {
                _Float16* dst = (p == 0) ? qs : ks;
#pragma unroll
                for (int i = 0; i < 8; ++i)
                    dst[(tm * 16 + 8 * half + i) * H_DIM + (tn * 16 + lw)] = (_Float16)c[i];
            }
        }
    }
    __syncthreads();

    // ---- phase 2: w = q k^T * H^-0.5 + grav(t), causal mask (16 tiles) ----
    const float scale = 0.17677669529663687f;   // 32^-0.5
    {
        const int tn  = wv & 3;      // both of this wave's tiles share tn
        const int tm0 = wv >> 2;     // 0..1
        v16h bb = load_b_h16(ks, H_DIM, tn * 16, 0, lw, half);
#pragma unroll
        for (int r = 0; r < 2; ++r) {
            int tm = tm0 + 2 * r;
            v16h a = load_a_h16(qs, H_DIM, tm * 16 + lw, 0, half);
            v8f c = {};
            c = wmma16(a, bb, c);
#pragma unroll
            for (int i = 0; i < 8; ++i) {
                int m = tm * 16 + 8 * half + i;
                int n = tn * 16 + lw;
                float val = c[i] * scale + gsh[m];
                wsm[m * T_DIM + n] = (n > m) ? -__builtin_inff() : val;
            }
        }
    }
    __syncthreads();

    // ---- phase 3: row softmax in f32, 4 threads per row (all 256 active) ----
    {
        const int row  = tid >> 2;           // 0..63
        const int part = tid & 3;            // 16 columns each
        float* rp = wsm + row * T_DIM + part * 16;

        float m = rp[0];
#pragma unroll
        for (int s = 1; s < 16; ++s) m = fmaxf(m, rp[s]);
        red2[tid] = m;
        __syncthreads();
        float rm = fmaxf(fmaxf(red2[row * 4 + 0], red2[row * 4 + 1]),
                         fmaxf(red2[row * 4 + 2], red2[row * 4 + 3]));
        float sum = 0.0f;
#pragma unroll
        for (int s = 0; s < 16; ++s) {
            float e = expf(rp[s] - rm);      // exp(-inf)=0 handles the mask
            rp[s] = e;
            sum += e;
        }
        __syncthreads();                     // all reads of red2 (max) done
        red2[tid] = sum;
        __syncthreads();
        float inv = 1.0f / (red2[row * 4 + 0] + red2[row * 4 + 1] +
                            red2[row * 4 + 2] + red2[row * 4 + 3]);
#pragma unroll
        for (int s = 0; s < 16; ++s) rp[s] *= inv;
    }
    __syncthreads();

    // ---- phase 4: out = w @ v   (8 tiles, 1 per wave, K=64) ----
    {
        const int tm = wv >> 1;      // 0..3
        const int tn = wv & 1;       // 0..1
        v8f c = {};
#pragma unroll
        for (int k0 = 0; k0 < T_DIM; k0 += 32) {
            v16h a  = load_a_f32(wsm, T_DIM, tm * 16 + lw, k0, half);
            v16h bb = load_b_h16(vts, T_DIM, tn * 16, k0, lw, half);
            c = wmma16(a, bb, c);
        }
        float* ob = out + (size_t)b * (T_DIM * H_DIM);
#pragma unroll
        for (int i = 0; i < 8; ++i)
            ob[(tm * 16 + 8 * half + i) * H_DIM + (tn * 16 + lw)] = c[i];
    }
}

extern "C" void kernel_launch(void* const* d_in, const int* in_sizes, int n_in,
                              void* d_out, int out_size, void* d_ws, size_t ws_size,
                              hipStream_t stream) {
    const float* x   = (const float*)d_in[0];
    const float* Wk  = (const float*)d_in[1];
    const float* Wq  = (const float*)d_in[2];
    const float* Wv  = (const float*)d_in[3];
    const float* tgt = (const float*)d_in[4];
    float* grav = (float*)d_ws;                 // 64 floats of scratch
    float* outp = (float*)d_out;

    grav_kernel<<<T_DIM, 256, 0, stream>>>(tgt, grav, in_sizes[4]);
    attn_kernel<<<4096, 256, 0, stream>>>(x, Wq, Wk, Wv, grav, outp);
}
